// Net_1795296329932
// MI455X (gfx1250) — matrix operand order repaired and checked by reading, not compiled, not run on hardware
//
#include <hip/hip_runtime.h>
#include <hip/hip_bf16.h>
#include <math.h>

typedef __attribute__((ext_vector_type(16))) _Float16     v16h;
typedef __attribute__((ext_vector_type(8)))  float        v8f;
typedef __attribute__((ext_vector_type(4)))  unsigned int u32x4;
typedef __attribute__((ext_vector_type(8)))  unsigned int u32x8;

#define D_FEAT 512
#define D_HID  64
#define N_CLS  40
#define NPAD   48
#define KS1    272   // LDS k-stride (per 256-K phase) for W1^T: 256 f16 + 16 f16 pad
#define KS2    80    // LDS k-stride for W2^T: 64 f16 + 16 f16 pad

// ---------------- TDM: DMA a 2D f16 tile Global->LDS with hardware padding ----
// rows    : tile_dim1 (# rows)
// rowElts : tile_dim0 (f16 elements per row; row byte size must be mult. of 4)
// strideE : tensor_dim0_stride in elements (global row stride)
// padIntvCode : pad every (1<<code) 8-byte units  (code 6 -> 512B, code 4 -> 128B)
// padAmtCode  : pad amount in DWORDs, encoded minus one (7 -> 8 dwords = 32B)
__device__ __forceinline__ void tdm_load_tile_f16(unsigned ldsOff,
                                                  unsigned long long gaddr,
                                                  unsigned rowElts, unsigned rows,
                                                  unsigned tensorD0, unsigned strideE,
                                                  unsigned padIntvCode,
                                                  unsigned padAmtCode) {
  u32x4 g0;
  g0[0] = 1u;                                        // count=1 (valid user D#)
  g0[1] = ldsOff;                                    // lds_addr (bytes)
  g0[2] = (unsigned)gaddr;                           // global_addr[31:0]
  g0[3] = (unsigned)((gaddr >> 32) & 0x01FFFFFFull)  // global_addr[56:32]
        | (2u << 30);                                // type = 2 ("image")
  u32x8 g1;
  g1[0] = (1u << 16)                                 // data_size = 2 bytes
        | (1u << 20)                                 // pad_enable
        | (padIntvCode << 22)
        | (padAmtCode << 25);
  g1[1] = (tensorD0 & 0xFFFFu) << 16;                // tensor_dim0[15:0]
  g1[2] = (tensorD0 >> 16) | ((rows & 0xFFFFu) << 16);   // dim0 hi | tensor_dim1 lo
  g1[3] = (rows >> 16) | (rowElts << 16);            // tensor_dim1 hi | tile_dim0
  g1[4] = rows & 0xFFFFu;                            // tile_dim1 (tile_dim2 = 0)
  g1[5] = strideE;                                   // tensor_dim0_stride[31:0]
  g1[6] = 0u;                                        // stride hi | dim1_stride lo
  g1[7] = 0u;
  u32x4 gz = {0u, 0u, 0u, 0u};                       // groups 2/3 unused (zero)
  asm volatile("tensor_load_to_lds %0, %1, %2, %3"
               :: "s"(g0), "s"(g1), "s"(gz), "s"(gz)
               : "memory");
}

// ---------------- norm build ----------------
__global__ void k_init_deg(float* deg, int n) {
  int i = blockIdx.x * 256 + threadIdx.x;
  if (i < n) deg[i] = 1.0f;  // self-loop weight
}

__global__ void k_deg_scatter(const long long* __restrict__ ei,
                              const float* __restrict__ ew,
                              float* deg, int E) {
  int e = blockIdx.x * 256 + threadIdx.x;
  if (e < E) atomicAdd(deg + ei[(size_t)E + e], ew[e]);
}

__global__ void k_dinv(const float* __restrict__ deg, float* dinv, int n) {
  int i = blockIdx.x * 256 + threadIdx.x;
  if (i < n) { float d = deg[i]; dinv[i] = (d > 0.f) ? rsqrtf(d) : 0.f; }
}

// ---------------- weight transpose/convert ----------------
__global__ void k_w1t(const float* __restrict__ W1, _Float16* __restrict__ w1t) {
  int i = blockIdx.x * 256 + threadIdx.x;          // [0, 64*512)
  if (i < D_HID * D_FEAT) {
    int n = i / D_FEAT, k = i % D_FEAT;
    w1t[i] = (_Float16)W1[(size_t)k * D_HID + n];  // W1 is [512][64]
  }
}

__global__ void k_w2t(const float* __restrict__ W2, _Float16* __restrict__ w2t) {
  int i = blockIdx.x * 256 + threadIdx.x;          // [0, 48*64)
  if (i < NPAD * D_HID) {
    int n = i / D_HID, k = i % D_HID;
    w2t[i] = (n < N_CLS) ? (_Float16)W2[(size_t)k * N_CLS + n] : (_Float16)0.f;
  }
}

// ---------------- GEMM1: h1 = x @ W1  (M=N nodes, K=512, N=64) ----------------
__global__ __launch_bounds__(256) void k_gemm1(const float* __restrict__ x,
                                               const _Float16* __restrict__ w1t,
                                               float* __restrict__ h1, int nNodes) {
  __shared__ _Float16 wlds[D_HID * KS1];  // 64 * 272 * 2B = 34,816 B
  const int tid  = threadIdx.x;
  const int wave = tid >> 5;
  const int lane = tid & 31;
  const int n0   = lane & 15;
  const int half = lane >> 4;
  const int waveRow = (blockIdx.x * 8 + wave) * 16;
  const bool active = (waveRow + 15) < nNodes;
  const unsigned ldsOff = (unsigned)(unsigned long long)(void*)wlds;

  v8f acc[4] = {};
  for (int phase = 0; phase < 2; ++phase) {
    __syncthreads();  // previous phase fully consumed before LDS overwrite
    if (wave == 0) {
      // DMA 64 rows x 256 f16 of W1^T into LDS; HW inserts 32B pad per 512B row
      tdm_load_tile_f16(ldsOff,
                        (unsigned long long)w1t + (unsigned long long)phase * 512u,
                        /*rowElts=*/256u, /*rows=*/D_HID,
                        /*tensorD0=*/D_FEAT, /*strideE=*/D_FEAT,
                        /*padIntvCode=*/6u /*512B*/, /*padAmtCode=*/7u /*8 dw*/);
      __builtin_amdgcn_s_wait_tensorcnt(0);
    }
    __syncthreads();

    if (active) {
      const float* xr = x + (size_t)(waveRow + n0) * D_FEAT + phase * 256;
      if (phase == 0) __builtin_prefetch(xr + 256, 0, 0);  // global_prefetch_b8
      for (int kt = 0; kt < 8; ++kt) {
        const int kb = kt * 32 + half * 16;
        // A operand: one contiguous 16-K run per lane (matches B)
        v16h a;
#pragma unroll
        for (int i = 0; i < 8; ++i) a[i] = (_Float16)xr[kb + i];
#pragma unroll
        for (int i = 0; i < 8; ++i) a[8 + i] = (_Float16)xr[kb + 8 + i];
#pragma unroll
        for (int nt = 0; nt < 4; ++nt) {
          const v16h b = *(const v16h*)(&wlds[(nt * 16 + n0) * KS1 + kt * 32 + half * 16]);
          acc[nt] = __builtin_amdgcn_wmma_f32_16x16x32_f16(
              false, a, false, b, (short)0, acc[nt], false, false);
        }
      }
    }
  }
  if (active) {
#pragma unroll
    for (int nt = 0; nt < 4; ++nt)
#pragma unroll
      for (int r = 0; r < 8; ++r) {
        int m = waveRow + half * 8 + r;          // C/D layout: lanes16-31 -> M=8+r
        h1[(size_t)m * D_HID + nt * 16 + n0] = acc[nt][r];
      }
  }
}

// ---------------- self-loop init: agg = lin * dinv^2 ----------------
__global__ void k_selfloop(const float* __restrict__ lin, const float* __restrict__ dinv,
                           float* __restrict__ agg, long long total, int feat) {
  long long i = (long long)blockIdx.x * 256 + threadIdx.x;
  if (i < total) {
    int node = (int)(i / feat);
    float d = dinv[node];
    agg[i] = lin[i] * d * d;
  }
}

// ---------------- edge scatter, layer 1 (64 feats, 2 per lane) ----------------
__global__ void k_edge1(const long long* __restrict__ ei, const float* __restrict__ ew,
                        const float* __restrict__ dinv, const float* __restrict__ h,
                        float* __restrict__ agg, int E) {
  int e = blockIdx.x * 8 + (threadIdx.x >> 5);
  if (e >= E) return;
  int lane = threadIdx.x & 31;
  long long r = ei[e];
  long long c = ei[(size_t)E + e];
  float w = ew[e] * dinv[r] * dinv[c];
  const float* hr = h + (size_t)r * D_HID;
  float* ac = agg + (size_t)c * D_HID;
  atomicAdd(ac + lane,      w * hr[lane]);
  atomicAdd(ac + lane + 32, w * hr[lane + 32]);
}

// ---------------- GEMM2: lin2 = relu(agg1+b1) @ W2  (K=64, N=40 pad 48) -------
__global__ __launch_bounds__(256) void k_gemm2(const float* __restrict__ agg1,
                                               const float* __restrict__ b1,
                                               const _Float16* __restrict__ w2t,
                                               float* __restrict__ lin2, int nNodes) {
  __shared__ _Float16 w2s[NPAD * KS2];  // 48 * 80 * 2B = 7,680 B
  __shared__ float    b1s[D_HID];
  const int tid  = threadIdx.x;
  const int wave = tid >> 5;
  const int lane = tid & 31;
  const int n0   = lane & 15;
  const int half = lane >> 4;
  const int waveRow = (blockIdx.x * 8 + wave) * 16;

  if (wave == 0) {
    // DMA 48 rows x 64 f16 of W2^T into LDS; HW inserts 32B pad per 128B row
    tdm_load_tile_f16((unsigned)(unsigned long long)(void*)w2s,
                      (unsigned long long)w2t,
                      /*rowElts=*/D_HID, /*rows=*/NPAD,
                      /*tensorD0=*/D_HID, /*strideE=*/D_HID,
                      /*padIntvCode=*/4u /*128B*/, /*padAmtCode=*/7u /*8 dw*/);
    __builtin_amdgcn_s_wait_tensorcnt(0);
  }
  if (tid < D_HID) b1s[tid] = b1[tid];
  __syncthreads();

  if ((waveRow + 15) >= nNodes) return;

  v8f acc[3] = {};
  const float* ar = agg1 + (size_t)(waveRow + n0) * D_HID;
#pragma unroll
  for (int kt = 0; kt < 2; ++kt) {
    const int kb = kt * 32 + half * 16;
    v16h a;
#pragma unroll
    for (int i = 0; i < 8; ++i)
      a[i] = (_Float16)fmaxf(ar[kb + i] + b1s[kb + i], 0.0f);
#pragma unroll
    for (int i = 0; i < 8; ++i)
      a[8 + i] = (_Float16)fmaxf(ar[kb + 8 + i] + b1s[kb + 8 + i], 0.0f);
#pragma unroll
    for (int nt = 0; nt < 3; ++nt) {
      const v16h b = *(const v16h*)(&w2s[(nt * 16 + n0) * KS2 + kt * 32 + half * 16]);
      acc[nt] = __builtin_amdgcn_wmma_f32_16x16x32_f16(
          false, a, false, b, (short)0, acc[nt], false, false);
    }
  }
#pragma unroll
  for (int nt = 0; nt < 3; ++nt) {
    int col = nt * 16 + n0;
    if (col < N_CLS) {
#pragma unroll
      for (int r = 0; r < 8; ++r) {
        int m = waveRow + half * 8 + r;
        lin2[(size_t)m * N_CLS + col] = acc[nt][r];
      }
    }
  }
}

// ---------------- edge scatter, layer 2 (40 feats) ----------------
__global__ void k_edge2(const long long* __restrict__ ei, const float* __restrict__ ew,
                        const float* __restrict__ dinv, const float* __restrict__ lin2,
                        float* __restrict__ agg, int E) {
  int e = blockIdx.x * 8 + (threadIdx.x >> 5);
  if (e >= E) return;
  int lane = threadIdx.x & 31;
  long long r = ei[e];
  long long c = ei[(size_t)E + e];
  float w = ew[e] * dinv[r] * dinv[c];
  const float* hr = lin2 + (size_t)r * N_CLS;
  float* ac = agg + (size_t)c * N_CLS;
  atomicAdd(ac + lane, w * hr[lane]);          // lanes 0..31 -> feats 0..31
  if (lane < N_CLS - 32)
    atomicAdd(ac + lane + 32, w * hr[lane + 32]);
}

// ---------------- final bias + log_softmax ----------------
__global__ void k_logsoftmax(const float* __restrict__ agg2, const float* __restrict__ b2,
                             float* __restrict__ out, int nNodes) {
  int i = blockIdx.x * 256 + threadIdx.x;
  if (i >= nNodes) return;
  const float* v = agg2 + (size_t)i * N_CLS;
  float vals[N_CLS];
  float m = -INFINITY;
#pragma unroll
  for (int c = 0; c < N_CLS; ++c) { vals[c] = v[c] + b2[c]; m = fmaxf(m, vals[c]); }
  float s = 0.f;
#pragma unroll
  for (int c = 0; c < N_CLS; ++c) s += expf(vals[c] - m);
  float ls = logf(s);
#pragma unroll
  for (int c = 0; c < N_CLS; ++c) out[(size_t)i * N_CLS + c] = vals[c] - m - ls;
}

extern "C" void kernel_launch(void* const* d_in, const int* in_sizes, int n_in,
                              void* d_out, int out_size, void* d_ws, size_t ws_size,
                              hipStream_t stream) {
  const float*     x  = (const float*)d_in[0];
  const long long* ei = (const long long*)d_in[1];
  const float*     ew = (const float*)d_in[2];
  const float*     W1 = (const float*)d_in[3];
  const float*     b1 = (const float*)d_in[4];
  const float*     W2 = (const float*)d_in[5];
  const float*     b2 = (const float*)d_in[6];
  float* out = (float*)d_out;

  const int N = in_sizes[0] / D_FEAT;   // 100000
  const int E = in_sizes[1] / 2;        // 3200000

  char* p = (char*)d_ws;
  auto alloc = [&](size_t bytes) {
    void* r = (void*)p;
    p += (bytes + 255) & ~(size_t)255;
    return r;
  };
  float*    deg  = (float*)alloc((size_t)N * 4);
  float*    dinv = (float*)alloc((size_t)N * 4);
  float*    h1   = (float*)alloc((size_t)N * D_HID * 4);
  float*    agg1 = (float*)alloc((size_t)N * D_HID * 4);
  float*    lin2 = (float*)alloc((size_t)N * N_CLS * 4);
  float*    agg2 = (float*)alloc((size_t)N * N_CLS * 4);
  _Float16* w1t  = (_Float16*)alloc((size_t)D_HID * D_FEAT * 2);
  _Float16* w2t  = (_Float16*)alloc((size_t)NPAD * D_HID * 2);

  // norm build
  k_init_deg   <<<(N + 255) / 256, 256, 0, stream>>>(deg, N);
  k_deg_scatter<<<(E + 255) / 256, 256, 0, stream>>>(ei, ew, deg, E);
  k_dinv       <<<(N + 255) / 256, 256, 0, stream>>>(deg, dinv, N);

  // weight prep (f16 transposed)
  k_w1t<<<(D_HID * D_FEAT + 255) / 256, 256, 0, stream>>>(W1, w1t);
  k_w2t<<<(NPAD * D_HID + 255) / 256, 256, 0, stream>>>(W2, w2t);

  const int waves = N / 16;                 // 6250 (N divisible by 16)
  const int gemmBlocks = (waves + 7) / 8;

  // layer 1
  k_gemm1<<<gemmBlocks, 256, 0, stream>>>(x, w1t, h1, N);
  {
    long long tot = (long long)N * D_HID;
    k_selfloop<<<(unsigned)((tot + 255) / 256), 256, 0, stream>>>(h1, dinv, agg1, tot, D_HID);
  }
  k_edge1<<<(E + 7) / 8, 256, 0, stream>>>(ei, ew, dinv, h1, agg1, E);

  // layer 2 (bias+ReLU fused into GEMM2 A-load)
  k_gemm2<<<gemmBlocks, 256, 0, stream>>>(agg1, b1, w2t, lin2, N);
  {
    long long tot = (long long)N * N_CLS;
    k_selfloop<<<(unsigned)((tot + 255) / 256), 256, 0, stream>>>(lin2, dinv, agg2, tot, N_CLS);
  }
  k_edge2<<<(E + 7) / 8, 256, 0, stream>>>(ei, ew, dinv, lin2, agg2, E);

  // output
  k_logsoftmax<<<(N + 255) / 256, 256, 0, stream>>>(agg2, b2, out, N);
}
